// CtrlPointDeformableTransformerDecoderLayer_14276471292332
// MI455X (gfx1250) — matrix operand order repaired and checked by reading, not compiled, not run on hardware
//
#include <hip/hip_runtime.h>
#include <hip/hip_bf16.h>
#include <math.h>

// ---------------------------------------------------------------------------
// Problem constants (static per the reference)
// ---------------------------------------------------------------------------
#define D_MODEL 256
#define DFFN_   1024
#define NHEAD   8
#define DHEAD   32
#define NLVL    4
#define NPT     4
#define BS_     8
#define NQ_     100
#define NPTS_   25
#define LQ_     (NQ_ * NPTS_)      // 2500
#define NTOK    (BS_ * LQ_)        // 20000 rows of 256
#define LEN_    21760              // 128*128 + 64*64 + 32*32 + 16*16
#define MVAL    (BS_ * LEN_)       // 174080 rows of 256

typedef __bf16 v16bf __attribute__((ext_vector_type(16)));
typedef float  v8f   __attribute__((ext_vector_type(8)));

// ---------------------------------------------------------------------------
// GEMM:  C[M,N] = act( A[M,K] @ W[N,K]^T + bias[N] )
// bf16 WMMA (v_wmma_f32_16x16x32_bf16), fp32 accumulate.
//  - 8 waves / block; wave w owns the 16x16 C tile at rows [mTile, mTile+16).
//  - The 16xK weight slab for this block's N-tile is loaded + converted to
//    packed bf16 in LDS ONCE per block (it was previously fetched 8x, once
//    per wave). B fragments then come from LDS via ds_load_b128.
//  - K is split into two independent accumulator chains (all K % 64 == 0) so
//    consecutive v_wmma ops have no RAW hazard and can pipeline with the
//    A-side v_cvt_pk_bf16_f32 work.
//  - M may not be a multiple of 128: inactive waves still run the barrier and
//    the K loop (A reads clamped to a valid row), only stores are predicated,
//    keeping EXEC all-1s around every WMMA (ISA requirement).
// ---------------------------------------------------------------------------
__global__ void __launch_bounds__(256)
gemm_bf16_wmma_kernel(const float* __restrict__ A, const float* __restrict__ W,
                      const float* __restrict__ bias, float* __restrict__ C,
                      int M, int N, int K, int relu)
{
    extern __shared__ __align__(16) unsigned lds_u[];   // [16][K/2] packed bf16x2
    const int lane  = threadIdx.x & 31;
    const int wave  = threadIdx.x >> 5;
    const int nBase = blockIdx.x * 16;
    const int mTile = (blockIdx.y * 8 + wave) * 16;

    // Cooperative load + f32->bf16 convert of the weight tile (all 256 thr).
    const int halfK = K >> 1;
    for (int i = threadIdx.x; i < 16 * halfK; i += 256) {
        const int r  = i / halfK;
        const int kk = i - r * halfK;
        const float2 f = ((const float2*)(W + (size_t)(nBase + r) * K))[kk];
        union { __bf16 h[2]; unsigned u; } pk;
        pk.h[0] = (__bf16)f.x;
        pk.h[1] = (__bf16)f.y;
        lds_u[r * halfK + kk] = pk.u;
    }
    __syncthreads();

    const int lh = lane >> 4;                 // 0 | 1  (K-half select)
    const int lm = lane & 15;                 // row (A) / col (B) in tile
    const bool active = (mTile < M);
    const float* __restrict__ arow = A + (size_t)((active ? mTile : 0) + lm) * K;
    const __bf16* __restrict__ wlds = (const __bf16*)lds_u + (size_t)lm * K;

    v8f acc0 = {}, acc1 = {};
    for (int kb = 0; kb < K; kb += 64) {
        // A fragments: 16x32 A tile in contiguous K order (same as B) — lane holds row lm, K pattern:
        //   a[j]   -> k = kb + lh*16 + j        (j = 0..7)
        //   a[j+8] -> k = kb + lh*16 + 8 + j
        v16bf a0, a1;
        const float* ap = arow + kb + lh * 16;
#pragma unroll
        for (int j = 0; j < 8; ++j) {
            a0[j]     = (__bf16)ap[j];
            a0[j + 8] = (__bf16)ap[8 + j];
            a1[j]     = (__bf16)ap[32 + j];
            a1[j + 8] = (__bf16)ap[40 + j];
        }
        if (kb + 64 < K) __builtin_prefetch(ap + 64, 0, 3);
        // B fragments from LDS: lane holds column lm; 16 contiguous K values,
        // lanes 0-15 -> K = kb..kb+15, lanes 16-31 -> K = kb+16..kb+31.
        const v16bf b0 = *(const v16bf*)(wlds + kb + lh * 16);
        const v16bf b1 = *(const v16bf*)(wlds + kb + 32 + lh * 16);
        acc0 = __builtin_amdgcn_wmma_f32_16x16x32_bf16(
            false, a0, false, b0, (short)0, acc0, false, false);
        acc1 = __builtin_amdgcn_wmma_f32_16x16x32_bf16(
            false, a1, false, b1, (short)0, acc1, false, false);
    }

    if (!active) return;
    // C/D layout: acc[r] -> row = mTile + lh*8 + r, col = nBase + lm
    const float bb = bias ? bias[nBase + lm] : 0.f;
#pragma unroll
    for (int r = 0; r < 8; ++r) {
        float v = acc0[r] + acc1[r] + bb;
        if (relu) v = fmaxf(v, 0.f);
        C[(size_t)(mTile + lh * 8 + r) * N + nBase + lm] = v;
    }
}

// ---------------------------------------------------------------------------
// Elementwise add (float4 vectorized):  c = a + b
// ---------------------------------------------------------------------------
__global__ void __launch_bounds__(256)
add_vec_kernel(const float* __restrict__ a, const float* __restrict__ b,
               float* __restrict__ c, int n4)
{
    const int i = blockIdx.x * blockDim.x + threadIdx.x;
    if (i >= n4) return;
    const float4 x = ((const float4*)a)[i];
    const float4 y = ((const float4*)b)[i];
    float4 z; z.x = x.x + y.x; z.y = x.y + y.y; z.z = x.z + y.z; z.w = x.w + y.w;
    ((float4*)c)[i] = z;
}

// ---------------------------------------------------------------------------
// Fused residual + LayerNorm over rows of width 256.
// One wave32 per row; lane owns 8 strided elements; __shfl_xor reductions.
// ---------------------------------------------------------------------------
__global__ void __launch_bounds__(256)
add_layernorm_kernel(const float* __restrict__ a, const float* __restrict__ b,
                     const float* __restrict__ g, const float* __restrict__ beta,
                     float* __restrict__ out, int rows)
{
    const int wave = threadIdx.x >> 5;
    const int lane = threadIdx.x & 31;
    const int row  = blockIdx.x * 8 + wave;
    if (row >= rows) return;
    const float* pa = a + (size_t)row * D_MODEL;
    const float* pb = b + (size_t)row * D_MODEL;

    float x[8];
    float s = 0.f;
#pragma unroll
    for (int j = 0; j < 8; ++j) {
        x[j] = pa[lane + 32 * j] + pb[lane + 32 * j];
        s += x[j];
    }
#pragma unroll
    for (int off = 16; off > 0; off >>= 1) s += __shfl_xor(s, off, 32);
    const float mean = s * (1.f / D_MODEL);

    float v = 0.f;
#pragma unroll
    for (int j = 0; j < 8; ++j) { const float d = x[j] - mean; v += d * d; }
#pragma unroll
    for (int off = 16; off > 0; off >>= 1) v += __shfl_xor(v, off, 32);
    const float inv = rsqrtf(v * (1.f / D_MODEL) + 1e-5f);

    float* po = out + (size_t)row * D_MODEL;
#pragma unroll
    for (int j = 0; j < 8; ++j) {
        const int c = lane + 32 * j;
        po[c] = (x[j] - mean) * inv * g[c] + beta[c];
    }
}

// ---------------------------------------------------------------------------
// Fused multi-head attention (dense), online softmax.
// Layout of Q/K/V/O: [Bp, S, NHEAD*DHEAD]; one block per (batch', head);
// K/V staged in LDS; one thread per query row.
// ---------------------------------------------------------------------------
__global__ void __launch_bounds__(128)
mha_online_softmax_kernel(const float* __restrict__ Q, const float* __restrict__ K,
                          const float* __restrict__ V, float* __restrict__ O, int S)
{
    const int bh = blockIdx.x;
    const int b  = bh / NHEAD;
    const int h  = bh % NHEAD;
    extern __shared__ float lds[];
    float* Ks = lds;
    float* Vs = lds + S * DHEAD;

    const size_t base = (size_t)b * S * D_MODEL + h * DHEAD;
    for (int i = threadIdx.x; i < S * DHEAD; i += blockDim.x) {
        const int s = i >> 5, d = i & 31;
        Ks[i] = K[base + (size_t)s * D_MODEL + d];
        Vs[i] = V[base + (size_t)s * D_MODEL + d];
    }
    __syncthreads();

    const int r = threadIdx.x;
    if (r >= S) return;

    float q[DHEAD];
#pragma unroll
    for (int d = 0; d < DHEAD; ++d) q[d] = Q[base + (size_t)r * D_MODEL + d];

    const float scale = 0.17677669529663687f;     // 1/sqrt(32)
    float m = -1e30f, l = 0.f, acc[DHEAD];
#pragma unroll
    for (int d = 0; d < DHEAD; ++d) acc[d] = 0.f;

    for (int s = 0; s < S; ++s) {
        float sc = 0.f;
#pragma unroll
        for (int d = 0; d < DHEAD; ++d) sc += q[d] * Ks[s * DHEAD + d];
        sc *= scale;
        const float mn = fmaxf(m, sc);
        const float f  = __expf(m - mn);
        const float w  = __expf(sc - mn);
        l = l * f + w;
#pragma unroll
        for (int d = 0; d < DHEAD; ++d) acc[d] = acc[d] * f + w * Vs[s * DHEAD + d];
        m = mn;
    }
    const float inv = 1.f / l;
#pragma unroll
    for (int d = 0; d < DHEAD; ++d) O[base + (size_t)r * D_MODEL + d] = acc[d] * inv;
}

// ---------------------------------------------------------------------------
// Transpose [bs, d1, d2, 256] -> [bs, d2, d1, 256]; one row per block (float4)
// ---------------------------------------------------------------------------
__global__ void __launch_bounds__(64)
transpose12_kernel(const float* __restrict__ in, float* __restrict__ out,
                   int d1, int d2)
{
    const int row = blockIdx.x;           // over bs*d1*d2 input rows
    const int j = row % d2;
    const int t = row / d2;
    const int i = t % d1;
    const int b = t / d1;
    const float4* src = (const float4*)(in + (size_t)row * D_MODEL);
    float4* dst = (float4*)(out + ((size_t)((b * d2 + j) * d1 + i)) * D_MODEL);
    dst[threadIdx.x] = src[threadIdx.x];
}

// ---------------------------------------------------------------------------
// In-place softmax over contiguous groups of 16 (attention weights, L*P=16)
// ---------------------------------------------------------------------------
__global__ void __launch_bounds__(256)
softmax16_kernel(float* __restrict__ x, int ngroups)
{
    const int g = blockIdx.x * blockDim.x + threadIdx.x;
    if (g >= ngroups) return;
    float* p = x + (size_t)g * 16;
    float e[16];
    float m = p[0];
#pragma unroll
    for (int i = 1; i < 16; ++i) m = fmaxf(m, p[i]);
    float s = 0.f;
#pragma unroll
    for (int i = 0; i < 16; ++i) { e[i] = __expf(p[i] - m); s += e[i]; }
    const float inv = 1.f / s;
#pragma unroll
    for (int i = 0; i < 16; ++i) p[i] = e[i] * inv;
}

// ---------------------------------------------------------------------------
// Deformable sampling. One wave per (b, lq, head); lane = channel (DHEAD==32).
// value: [bs, LEN, 256] (head-major cols) ; off: [bs,Lq,256] ; aw: [bs,Lq,128]
// ref: [bs, Lq, 4, 2] ; out: [bs, Lq, 256]
// Every bilinear corner is a 128B coalesced gather; coords are wave-uniform.
// ---------------------------------------------------------------------------
__global__ void __launch_bounds__(256)
deform_sample_kernel(const float* __restrict__ value, const float* __restrict__ ref,
                     const float* __restrict__ off, const float* __restrict__ aw,
                     float* __restrict__ out)
{
    const int LVL_H[NLVL] = {128, 64, 32, 16};
    const int LVL_W[NLVL] = {128, 64, 32, 16};
    const int LVL_S[NLVL] = {0, 16384, 20480, 21504};

    const int g    = blockIdx.x * 8 + (threadIdx.x >> 5);
    const int lane = threadIdx.x & 31;
    if (g >= BS_ * LQ_ * NHEAD) return;
    const int h  = g % NHEAD;
    const int t  = g / NHEAD;
    const int lq = t % LQ_;
    const int b  = t / LQ_;

    const size_t rowq = (size_t)b * LQ_ + lq;
    const float* refp = ref + rowq * (NLVL * 2);
    const float* offp = off + rowq * D_MODEL + h * (NLVL * NPT * 2);
    const float* awp  = aw  + rowq * (NHEAD * 16) + h * 16;
    const size_t vch  = (size_t)h * DHEAD + lane;

    float acc = 0.f;
#pragma unroll
    for (int l = 0; l < NLVL; ++l) {
        const int   wl = LVL_W[l], hl = LVL_H[l];
        const float rx = refp[l * 2 + 0], ry = refp[l * 2 + 1];
        const size_t lbase = ((size_t)b * LEN_ + LVL_S[l]) * D_MODEL + vch;
#pragma unroll
        for (int p = 0; p < NPT; ++p) {
            const float ox = offp[(l * NPT + p) * 2 + 0];
            const float oy = offp[(l * NPT + p) * 2 + 1];
            const float x  = (rx + ox / (float)wl) * wl - 0.5f;
            const float y  = (ry + oy / (float)hl) * hl - 0.5f;
            const float xf = floorf(x), yf = floorf(y);
            const float wx = x - xf,    wy = y - yf;
            const int   x0 = (int)xf,   y0 = (int)yf;

            auto fetch = [&](int ix, int iy) -> float {
                if ((unsigned)ix >= (unsigned)wl || (unsigned)iy >= (unsigned)hl)
                    return 0.f;
                return value[lbase + (size_t)(iy * wl + ix) * D_MODEL];
            };
            const float s00 = fetch(x0,     y0);
            const float s10 = fetch(x0 + 1, y0);
            const float s01 = fetch(x0,     y0 + 1);
            const float s11 = fetch(x0 + 1, y0 + 1);
            const float bil = (s00 * (1.f - wx) + s10 * wx) * (1.f - wy)
                            + (s01 * (1.f - wx) + s11 * wx) * wy;
            acc += awp[l * NPT + p] * bil;
        }
    }
    out[rowq * D_MODEL + h * DHEAD + lane] = acc;
}

// ---------------------------------------------------------------------------
// Host-side launch helpers
// ---------------------------------------------------------------------------
static inline void launch_gemm(const float* A, const float* W, const float* bias,
                               float* C, int M, int N, int K, int relu,
                               hipStream_t s)
{
    dim3 grid(N / 16, (M + 127) / 128);
    const size_t lds_bytes = (size_t)32 * K;   // 16 rows * K/2 * sizeof(u32)
    gemm_bf16_wmma_kernel<<<grid, 256, lds_bytes, s>>>(A, W, bias, C, M, N, K, relu);
}

static inline void launch_add(const float* a, const float* b, float* c, int n,
                              hipStream_t s)
{
    const int n4 = n / 4;
    add_vec_kernel<<<(n4 + 255) / 256, 256, 0, s>>>(a, b, c, n4);
}

static inline void launch_ln(const float* a, const float* b, const float* g,
                             const float* beta, float* out, int rows,
                             hipStream_t s)
{
    add_layernorm_kernel<<<(rows + 7) / 8, 256, 0, s>>>(a, b, g, beta, out, rows);
}

// ---------------------------------------------------------------------------
// kernel_launch
// Input order (setup_inputs dict order, params dict flattened in insertion
// order): 0 tgt, 1 query_pos, 2 reference_points, 3 src, 4 src_spatial_shapes,
// 5 level_start_index, then params: 6 intra_in_w, 7 intra_in_b, 8 intra_out_w,
// 9 intra_out_b, 10 inter_in_w, 11 inter_in_b, 12 inter_out_w, 13 inter_out_b,
// 14 off_w, 15 off_b, 16 aw_w, 17 aw_b, 18 v_w, 19 v_b, 20 out_w, 21 out_b,
// 22 lin1_w, 23 lin1_b, 24 lin2_w, 25 lin2_b, 26..33 ln_{intra,inter,cross,3}_{g,b}
// Workspace requirement: 5*NTOK*256 + NTOK*128 + MVAL*256 floats (~291 MB).
// ---------------------------------------------------------------------------
extern "C" void kernel_launch(void* const* d_in, const int* in_sizes, int n_in,
                              void* d_out, int out_size, void* d_ws, size_t ws_size,
                              hipStream_t stream)
{
    (void)in_sizes; (void)n_in; (void)out_size; (void)ws_size;

    const float* tgt  = (const float*)d_in[0];
    const float* qpos = (const float*)d_in[1];
    const float* refp = (const float*)d_in[2];
    const float* src  = (const float*)d_in[3];

    const float* intra_in_w  = (const float*)d_in[6];
    const float* intra_in_b  = (const float*)d_in[7];
    const float* intra_out_w = (const float*)d_in[8];
    const float* intra_out_b = (const float*)d_in[9];
    const float* inter_in_w  = (const float*)d_in[10];
    const float* inter_in_b  = (const float*)d_in[11];
    const float* inter_out_w = (const float*)d_in[12];
    const float* inter_out_b = (const float*)d_in[13];
    const float* off_w  = (const float*)d_in[14];
    const float* off_b  = (const float*)d_in[15];
    const float* aw_w   = (const float*)d_in[16];
    const float* aw_b   = (const float*)d_in[17];
    const float* v_w    = (const float*)d_in[18];
    const float* v_b    = (const float*)d_in[19];
    const float* out_w  = (const float*)d_in[20];
    const float* out_b  = (const float*)d_in[21];
    const float* lin1_w = (const float*)d_in[22];
    const float* lin1_b = (const float*)d_in[23];
    const float* lin2_w = (const float*)d_in[24];
    const float* lin2_b = (const float*)d_in[25];
    const float* ln_intra_g = (const float*)d_in[26];
    const float* ln_intra_b = (const float*)d_in[27];
    const float* ln_inter_g = (const float*)d_in[28];
    const float* ln_inter_b = (const float*)d_in[29];
    const float* ln_cross_g = (const float*)d_in[30];
    const float* ln_cross_b = (const float*)d_in[31];
    const float* ln_3_g     = (const float*)d_in[32];
    const float* ln_3_b     = (const float*)d_in[33];

    // workspace bump allocation (floats)
    const size_t NT = (size_t)NTOK * D_MODEL;          // 5,120,000
    float* ws  = (float*)d_ws;
    float* X0  = ws;
    float* X1  = X0 + NT;
    float* X2  = X1 + NT;
    float* X3  = X2 + NT;
    float* X4  = X3 + NT;
    float* AWB = X4 + NT;                              // NTOK * 128
    float* BIG = AWB + (size_t)NTOK * 128;             // max(MVAL*256, NTOK*1024)

    // ---- Step A: intra-instance attention over control points (S=25) ----
    launch_add(tgt, qpos, X0, NTOK * D_MODEL, stream);                        // q1
    launch_gemm(X0,  intra_in_w,               intra_in_b,       X1, NTOK, 256, 256, 0, stream); // Q
    launch_gemm(X0,  intra_in_w + 256 * 256,   intra_in_b + 256, X2, NTOK, 256, 256, 0, stream); // K
    launch_gemm(tgt, intra_in_w + 512 * 256,   intra_in_b + 512, X3, NTOK, 256, 256, 0, stream); // V (from tgt)
    mha_online_softmax_kernel<<<BS_ * NQ_ * NHEAD, 128,
        NPTS_ * DHEAD * 2 * sizeof(float), stream>>>(X1, X2, X3, X4, NPTS_);  // O
    launch_gemm(X4, intra_out_w, intra_out_b, X0, NTOK, 256, 256, 0, stream); // t2
    launch_ln(tgt, X0, ln_intra_g, ln_intra_b, X1, NTOK, stream);             // T1

    // ---- Step B: inter-instance attention over queries (S=100) ----
    transpose12_kernel<<<NTOK, 64, 0, stream>>>(X1, X2, NQ_, NPTS_);          // TI [bs,25,100,256]
    launch_gemm(X2, inter_in_w,             inter_in_b,       X0, NTOK, 256, 256, 0, stream); // Qi
    launch_gemm(X2, inter_in_w + 256 * 256, inter_in_b + 256, X3, NTOK, 256, 256, 0, stream); // Ki
    launch_gemm(X2, inter_in_w + 512 * 256, inter_in_b + 512, X4, NTOK, 256, 256, 0, stream); // Vi
    mha_online_softmax_kernel<<<BS_ * NPTS_ * NHEAD, 128,
        NQ_ * DHEAD * 2 * sizeof(float), stream>>>(X0, X3, X4, X1, NQ_);      // Oi
    launch_gemm(X1, inter_out_w, inter_out_b, X0, NTOK, 256, 256, 0, stream); // t2i
    launch_ln(X2, X0, ln_inter_g, ln_inter_b, X3, NTOK, stream);              // TI2
    transpose12_kernel<<<NTOK, 64, 0, stream>>>(X3, X1, NPTS_, NQ_);          // tgt_inter [bs,100,25,256]

    // ---- Step C: deformable cross-attention ----
    launch_add(X1, qpos, X0, NTOK * D_MODEL, stream);                         // query
    launch_gemm(src, v_w,   v_b,   BIG, MVAL, 256, 256, 0, stream);           // value [174080,256]
    launch_gemm(X0,  off_w, off_b, X2,  NTOK, 256, 256, 0, stream);           // offsets
    launch_gemm(X0,  aw_w,  aw_b,  AWB, NTOK, 128, 256, 0, stream);           // aw logits
    softmax16_kernel<<<(NTOK * NHEAD + 255) / 256, 256, 0, stream>>>(
        AWB, NTOK * NHEAD);                                                   // softmax over L*P
    deform_sample_kernel<<<NTOK, 256, 0, stream>>>(BIG, refp, X2, AWB, X3);   // sampled
    launch_gemm(X3, out_w, out_b, X0, NTOK, 256, 256, 0, stream);             // t2c
    launch_ln(X1, X0, ln_cross_g, ln_cross_b, X2, NTOK, stream);              // tgt2

    // ---- Step D: FFN + final LN ----
    launch_gemm(X2,  lin1_w, lin1_b, BIG, NTOK, 1024, 256, 1, stream);        // relu(lin1)
    launch_gemm(BIG, lin2_w, lin2_b, X0,  NTOK, 256, 1024, 0, stream);        // lin2
    launch_ln(X2, X0, ln_3_g, ln_3_b, (float*)d_out, NTOK, stream);           // output
}
